// AttnBlock_13469017440567
// MI455X (gfx1250) — matrix operand order repaired and checked by reading, not compiled, not run on hardware
//
#include <hip/hip_runtime.h>

// ---------------------------------------------------------------------------
// AttnBlock for MI455X (gfx1250): bf16 WMMA everywhere, flash-style softmax,
// async global->LDS staging (ASYNCcnt) for the QKV input slab.
// ---------------------------------------------------------------------------

#define BATCH 32
#define CH    256
#define NTOK  1024
#define NH    4
#define DKH   64
#define O3    768           // 3 * NH * DKH
#define ATT_SCALE 0.125f    // 1/sqrt(64)
#define XPITCH 260          // 256 + 4 dword pad: A-frag rows hit 16 distinct banks

typedef __bf16 bf16_t;
typedef __attribute__((ext_vector_type(16))) __bf16 v16bf;
typedef __attribute__((ext_vector_type(8)))  __bf16 v8bf;
typedef __attribute__((ext_vector_type(8)))  float  v8f;
typedef __attribute__((ext_vector_type(4)))  float  v4f;

#define WMMA_BF16(A, B, C) \
  __builtin_amdgcn_wmma_f32_16x16x32_bf16(false, (A), false, (B), (short)0, (C), false, false)

// Build a 16x32 bf16 A-fragment from two 16-byte bf16 chunks (global or LDS).
// k order used here (contiguous, same as the B-frag loads): lane L holds row L%16; chunk0 = K
// [ (L/16)*16 .. +7 ], chunk1 = K [ (L/16)*16 + 8 .. +7 ].
static __device__ __forceinline__ v16bf make_a_frag(const bf16_t* lo_p, const bf16_t* hi_p) {
  v8bf lo = *(const v8bf*)lo_p;
  v8bf hi = *(const v8bf*)hi_p;
  v16bf a;
#pragma unroll
  for (int q = 0; q < 8; ++q) { a[q] = lo[q]; a[8 + q] = hi[q]; }
  return a;
}

// Same fragment, sourced from fp32 LDS (convert during fragment build;
// lowers to v_cvt_pk_bf16_f32 pairs).
static __device__ __forceinline__ v16bf make_a_frag_f32(const float* lo_p, const float* hi_p) {
  v4f l0 = *(const v4f*)lo_p, l1 = *(const v4f*)(lo_p + 4);
  v4f h0 = *(const v4f*)hi_p, h1 = *(const v4f*)(hi_p + 4);
  v16bf a;
#pragma unroll
  for (int q = 0; q < 4; ++q) {
    a[q]      = (bf16_t)l0[q];
    a[4 + q]  = (bf16_t)l1[q];
    a[8 + q]  = (bf16_t)h0[q];
    a[12 + q] = (bf16_t)h1[q];
  }
  return a;
}

// ---------------------------------------------------------------------------
// Kernel 0: fp32 -> bf16 weight conversion
// ---------------------------------------------------------------------------
__global__ void cvt_kernel(const float* __restrict__ src, bf16_t* __restrict__ dst, int n) {
  int i = blockIdx.x * 256 + threadIdx.x;
  if (i < n) dst[i] = (bf16_t)src[i];
}

// ---------------------------------------------------------------------------
// Kernel 1: fused QKV projection.  qkv[n][o] = sum_c x[b][c][n] * pw[o][c] + pb[o]
// Block = (b, 16-token tile). The whole 16x256 (strided) x-slab is staged to
// LDS once via GLOBAL_LOAD_ASYNC_TO_LDS_B32 (no VGPR round-trip, ASYNCcnt),
// one barrier pair, then 8 waves x 6 o-tiles x 8 K-steps of WMMA.
// Q,K stored [b,h,n,64] row-major bf16; V stored transposed [b,h,64,n].
// ---------------------------------------------------------------------------
__global__ __launch_bounds__(256) void qkv_kernel(const float* __restrict__ x,
                                                  const bf16_t* __restrict__ pw,
                                                  const float* __restrict__ pb,
                                                  bf16_t* __restrict__ Qt,
                                                  bf16_t* __restrict__ Kt,
                                                  bf16_t* __restrict__ Vt) {
  __shared__ __align__(16) float lds_x[16 * XPITCH];   // [n][c] fp32, padded rows

  const int b   = blockIdx.x >> 6;
  const int n0  = (blockIdx.x & 63) << 4;
  const int tid  = threadIdx.x;
  const int lane = tid & 31;
  const int wave = tid >> 5;
  const int lhi  = lane >> 4;
  const int llo  = lane & 15;

  // ---- async stage: x[b][0..255][n0..n0+15] -> LDS[nl][c] ------------------
  const float* xsrc = x + (size_t)b * CH * NTOK + n0;  // element (c,nl) @ c*NTOK+nl
  const unsigned lds0 = (unsigned)(uintptr_t)&lds_x[0];
#pragma unroll
  for (int e = tid; e < 4096; e += 256) {
    int c = e >> 4, nl = e & 15;
    unsigned loff = lds0 + (unsigned)((nl * XPITCH + c) * 4);
    unsigned goff = (unsigned)((c * NTOK + nl) * 4);
    asm volatile("global_load_async_to_lds_b32 %0, %1, %2"
                 :: "v"(loff), "v"(goff), "s"(xsrc)
                 : "memory");
  }
  asm volatile("s_wait_asynccnt 0" ::: "memory");
  __syncthreads();

  // ---- GEMM: 6 o-tiles per wave, K loop over C=256 in steps of 32 ----------
  v8f acc[6] = {};
  for (int ks = 0; ks < 8; ++ks) {
    const int c0 = ks * 32;
    v16bf a = make_a_frag_f32(&lds_x[llo * XPITCH + c0 + lhi * 16],
                              &lds_x[llo * XPITCH + c0 + 8 + lhi * 16]);
#pragma unroll
    for (int t = 0; t < 6; ++t) {
      int o = ((wave * 6 + t) << 4) + llo;   // B-frag: col = lane%16, K chunk by lane/16
      v16bf bfrag = *(const v16bf*)&pw[(size_t)o * CH + c0 + lhi * 16];
      acc[t] = WMMA_BF16(a, bfrag, acc[t]);
    }
  }

  // ---- epilogue: bias + scatter into Q / K / V^T ---------------------------
#pragma unroll
  for (int t = 0; t < 6; ++t) {
    int o   = ((wave * 6 + t) << 4) + llo;
    float bias = pb[o];
    int h   = o / 192;
    int rem = o % 192;
    size_t bh = (size_t)(b * NH + h);
#pragma unroll
    for (int r = 0; r < 8; ++r) {            // C layout: row = r + 8*(lane/16)
      int n = n0 + r + lhi * 8;
      bf16_t hv = (bf16_t)(acc[t][r] + bias);
      if (rem < 64)       Qt[(bh * NTOK + n) * DKH + rem] = hv;
      else if (rem < 128) Kt[(bh * NTOK + n) * DKH + (rem - 64)] = hv;
      else                Vt[(bh * DKH + (rem - 128)) * NTOK + n] = hv;
    }
  }
}

// ---------------------------------------------------------------------------
// Kernel 2: flash attention per (b, h, 128-query block); wave = 16 queries.
// Online softmax: register row-max/row-sum via __shfl_xor within 16-lane halves.
// ---------------------------------------------------------------------------
__global__ __launch_bounds__(256) void attn_kernel(const bf16_t* __restrict__ Qt,
                                                   const bf16_t* __restrict__ Kt,
                                                   const bf16_t* __restrict__ Vt,
                                                   bf16_t* __restrict__ ctx) {
  __shared__ __align__(32) bf16_t lds_p[8 * 16 * 32];   // per-wave P relayout scratch

  const int bh   = blockIdx.x >> 3;          // b*NH + h
  const int b    = bh >> 2, h = bh & 3;
  const int iblk = blockIdx.x & 7;
  const int tid  = threadIdx.x;
  const int lane = tid & 31;
  const int wave = tid >> 5;
  const int lhi  = lane >> 4;
  const int llo  = lane & 15;
  const int i0   = iblk * 128 + wave * 16;

  const bf16_t* Qb = Qt + (size_t)bh * NTOK * DKH;
  const bf16_t* Kb = Kt + (size_t)bh * NTOK * DKH;
  const bf16_t* Vb = Vt + (size_t)bh * DKH * NTOK;
  bf16_t* myp = &lds_p[wave * 512];

  // Q fragments (rows i0..i0+15, d split 0..31 / 32..63), kept in registers
  v16bf aq[2];
#pragma unroll
  for (int ks = 0; ks < 2; ++ks) {
    const bf16_t* qrow = Qb + (size_t)(i0 + llo) * DKH + ks * 32;
    aq[ks] = make_a_frag(qrow + lhi * 16, qrow + 8 + lhi * 16);
  }

  v8f acc[4] = {};
  float m[8], l[8];
#pragma unroll
  for (int r = 0; r < 8; ++r) { m[r] = -1e30f; l[r] = 0.0f; }

  for (int j0 = 0; j0 < NTOK; j0 += 32) {
    if (j0 + 32 < NTOK)
      __builtin_prefetch(Kb + (size_t)(j0 + 32 + llo) * DKH, 0, 0);

    // S = (Q K^T) * scale : two 16x16 n-tiles, K(=d) loop of 2
    v8f s[2] = {};
#pragma unroll
    for (int t = 0; t < 2; ++t) {
      int j = j0 + t * 16 + llo;
#pragma unroll
      for (int ks = 0; ks < 2; ++ks) {
        v16bf kb = *(const v16bf*)(Kb + (size_t)j * DKH + ks * 32 + lhi * 16);
        s[t] = WMMA_BF16(aq[ks], kb, s[t]);
      }
    }

    // online softmax (rows striped: row = r + 8*(lane/16), cols across 16 lanes)
    float rowmax[8], corr[8], rowsum[8];
#pragma unroll
    for (int r = 0; r < 8; ++r) {
      float a0 = s[0][r] * ATT_SCALE, a1 = s[1][r] * ATT_SCALE;
      s[0][r] = a0; s[1][r] = a1;
      rowmax[r] = fmaxf(a0, a1);
    }
#pragma unroll
    for (int r = 0; r < 8; ++r)
#pragma unroll
      for (int off = 1; off < 16; off <<= 1)
        rowmax[r] = fmaxf(rowmax[r], __shfl_xor(rowmax[r], off, 32));
#pragma unroll
    for (int r = 0; r < 8; ++r) {
      float mn = fmaxf(m[r], rowmax[r]);
      corr[r]  = __expf(m[r] - mn);
      m[r]     = mn;
      float p0 = __expf(s[0][r] - mn);
      float p1 = __expf(s[1][r] - mn);
      s[0][r] = p0; s[1][r] = p1;
      rowsum[r] = p0 + p1;
    }
#pragma unroll
    for (int r = 0; r < 8; ++r)
#pragma unroll
      for (int off = 1; off < 16; off <<= 1)
        rowsum[r] += __shfl_xor(rowsum[r], off, 32);
#pragma unroll
    for (int r = 0; r < 8; ++r) l[r] = l[r] * corr[r] + rowsum[r];
#pragma unroll
    for (int d = 0; d < 4; ++d)
#pragma unroll
      for (int r = 0; r < 8; ++r) acc[d][r] *= corr[r];

    // P: C-layout regs -> LDS [16m][32j] -> A-fragment (same-wave DS, in-order)
#pragma unroll
    for (int t = 0; t < 2; ++t)
#pragma unroll
      for (int r = 0; r < 8; ++r)
        myp[(r + lhi * 8) * 32 + t * 16 + llo] = (bf16_t)s[t][r];
    v16bf pa = make_a_frag(&myp[llo * 32 + lhi * 16],
                           &myp[llo * 32 + 8 + lhi * 16]);

    // ctx += P @ V ; V^T gives contiguous B-fragment loads
#pragma unroll
    for (int d = 0; d < 4; ++d) {
      v16bf vb = *(const v16bf*)(Vb + (size_t)(d * 16 + llo) * NTOK + j0 + lhi * 16);
      acc[d] = WMMA_BF16(pa, vb, acc[d]);
    }
  }

  // normalize and store ctx[b][n][h*64+d] row-major bf16
#pragma unroll
  for (int d = 0; d < 4; ++d)
#pragma unroll
    for (int r = 0; r < 8; ++r) {
      int n = i0 + r + lhi * 8;
      int e = h * DKH + d * 16 + llo;
      ctx[((size_t)b * NTOK + n) * CH + e] = (bf16_t)(acc[d][r] / l[r]);
    }
}

// ---------------------------------------------------------------------------
// Kernel 3: out = ctx @ out_w^T + out_b + xs, stored [b][c][n] fp32.
// Wave = one 16x16 tile; A-fragments load directly from row-major ctx.
// ---------------------------------------------------------------------------
__global__ __launch_bounds__(256) void outproj_kernel(const bf16_t* __restrict__ ctx,
                                                      const bf16_t* __restrict__ ow,
                                                      const float* __restrict__ ob,
                                                      const float* __restrict__ x,
                                                      float* __restrict__ out) {
  const int tid  = threadIdx.x;
  const int lane = tid & 31;
  const int wave = tid >> 5;
  const int lhi  = lane >> 4;
  const int llo  = lane & 15;

  const int blk   = blockIdx.x;
  const int b     = blk >> 7;
  const int ntile = (blk >> 1) & 63;
  const int n0    = ntile << 4;
  const int c0    = ((((blk & 1) << 3) + wave) << 4);

  v8f acc = {};
  const bf16_t* arow = ctx + ((size_t)b * NTOK + n0 + llo) * CH;
  const bf16_t* brow = ow + (size_t)(c0 + llo) * CH;
#pragma unroll
  for (int ks = 0; ks < 8; ++ks) {
    int e0 = ks * 32;
    v16bf a = make_a_frag(arow + e0 + lhi * 16, arow + e0 + 8 + lhi * 16);
    v16bf bb = *(const v16bf*)(brow + e0 + lhi * 16);
    acc = WMMA_BF16(a, bb, acc);
  }

  const int c = c0 + llo;
  const float bias = ob[c];
  const float* xr = x + ((size_t)b * CH + c) * NTOK + n0 + lhi * 8;
  float* orow     = out + ((size_t)b * CH + c) * NTOK + n0 + lhi * 8;
  v4f r0 = *(const v4f*)xr;
  v4f r1 = *(const v4f*)(xr + 4);
  v4f o0, o1;
#pragma unroll
  for (int q = 0; q < 4; ++q) {
    o0[q] = acc[q]     + bias + r0[q];
    o1[q] = acc[4 + q] + bias + r1[q];
  }
  *(v4f*)orow       = o0;
  *(v4f*)(orow + 4) = o1;
}

// ---------------------------------------------------------------------------
// host-side launcher
// ---------------------------------------------------------------------------
extern "C" void kernel_launch(void* const* d_in, const int* in_sizes, int n_in,
                              void* d_out, int out_size, void* d_ws, size_t ws_size,
                              hipStream_t stream) {
  const float* x      = (const float*)d_in[0];
  const float* proj_w = (const float*)d_in[1];
  const float* proj_b = (const float*)d_in[2];
  const float* out_w  = (const float*)d_in[3];
  const float* out_b  = (const float*)d_in[4];
  float* out = (float*)d_out;

  // workspace layout (~64.5 MB)
  char* ws = (char*)d_ws;
  size_t off = 0;
  bf16_t* pw_h = (bf16_t*)(ws + off); off += (size_t)O3 * CH * 2;                 // 384 KB
  bf16_t* ow_h = (bf16_t*)(ws + off); off += (size_t)CH * CH * 2;                 // 128 KB
  bf16_t* Qt   = (bf16_t*)(ws + off); off += (size_t)BATCH * NH * NTOK * DKH * 2; // 16 MB
  bf16_t* Kt   = (bf16_t*)(ws + off); off += (size_t)BATCH * NH * NTOK * DKH * 2; // 16 MB
  bf16_t* Vt   = (bf16_t*)(ws + off); off += (size_t)BATCH * NH * DKH * NTOK * 2; // 16 MB
  bf16_t* ctx  = (bf16_t*)(ws + off); off += (size_t)BATCH * NTOK * CH * 2;       // 16 MB
  (void)ws_size; (void)in_sizes; (void)n_in; (void)out_size;

  cvt_kernel<<<(O3 * CH + 255) / 256, 256, 0, stream>>>(proj_w, pw_h, O3 * CH);
  cvt_kernel<<<(CH * CH + 255) / 256, 256, 0, stream>>>(out_w, ow_h, CH * CH);
  qkv_kernel<<<BATCH * 64, 256, 0, stream>>>(x, pw_h, proj_b, Qt, Kt, Vt);
  attn_kernel<<<BATCH * NH * 8, 256, 0, stream>>>(Qt, Kt, Vt, ctx);
  outproj_kernel<<<BATCH * 128, 256, 0, stream>>>(ctx, ow_h, out_b, x, out);
}
